// Speller_86174223827463
// MI455X (gfx1250) — compile-verified, matrix operand order checked by reading, not run on hardware
//
#include <hip/hip_runtime.h>
#include <math.h>

// ---------------------------------------------------------------------------
// Problem constants (from reference)
// ---------------------------------------------------------------------------
#define B_SZ   64
#define ENC_T  1024
#define DEC_T  256
#define UNITS  512
#define VOCAB  46

#define IN0    1071            // h0(512) + y(1) + context(512) + dist(46)
#define K0PAD  1600            // IN0 + UNITS (=1583) padded to multiple of 32
#define K1     1024            // context(512) + h1(512)
#define N_GATE 2048            // 4*UNITS
#define VPAD   48              // VOCAB padded to multiple of 16

typedef __attribute__((ext_vector_type(16))) _Float16 v16h;
typedef __attribute__((ext_vector_type(8)))  _Float16 v8h;
typedef __attribute__((ext_vector_type(8)))  float    v8f;

// ---------------------------------------------------------------------------
// Shared WMMA K-loop: one wave accumulates a 16x16 f32 tile.
//   A:  M x K f16 row-major (activations)
//   Bm: N x K f16 row-major (weights pre-transposed, i.e. column-major KxN)
// Fragment layout per CDNA5 ISA 7.12.2 (wave32, 16-bit A/B): lane l holds
// row/col (l&15); K-halves {0..7,16..23} vs {8..15,24..31} split on l>=16.
// ---------------------------------------------------------------------------
__device__ __forceinline__ v8f
wmma_loop(const _Float16* __restrict__ A, const _Float16* __restrict__ Bm,
          int K, int mtile, int ntile, int lane)
{
    const int l15  = lane & 15;
    const int koff = ((lane >> 4) & 1) * 8;
    const _Float16* __restrict__ arow = A  + (size_t)(mtile * 16 + l15) * K;
    const _Float16* __restrict__ brow = Bm + (size_t)(ntile * 16 + l15) * K;

    v8f acc = {};
    for (int kb = 0; kb < K; kb += 32) {
        __builtin_prefetch((const void*)(brow + kb + 256), 0, 1);  // global_prefetch_b8
        v8h a_lo = *(const v8h*)(arow + kb + koff);
        v8h a_hi = *(const v8h*)(arow + kb + 16 + koff);
        v8h b_lo = *(const v8h*)(brow + kb + koff);
        v8h b_hi = *(const v8h*)(brow + kb + 16 + koff);
        v16h av, bv;
#pragma unroll
        for (int i = 0; i < 8; ++i) {
            av[i] = a_lo[i]; av[8 + i] = a_hi[i];
            bv[i] = b_lo[i]; bv[8 + i] = b_hi[i];
        }
        acc = __builtin_amdgcn_wmma_f32_16x16x32_f16(
                  false, av, false, bv, (short)0, acc, false, false);
    }
    return acc;
}

// ---------------------------------------------------------------------------
// Fused WMMA GEMM + LSTM gates. Weight columns are gate-interleaved
// (n' = u*4 + gate), so one 16x16 tile = 16 batch rows x 4 units x 4 gates,
// entirely within one wave. Tile is transposed through 1KB of LDS, then the
// wave applies sigmoid/tanh and updates h/c in place (no z round-trip).
// ---------------------------------------------------------------------------
__global__ void __launch_bounds__(32)
wmma_lstm_kernel(const _Float16* __restrict__ A,
                 const _Float16* __restrict__ Bm,
                 const float* __restrict__ bias,       // original layout (4*U)
                 float* __restrict__ hstate,
                 float* __restrict__ cstate,
                 _Float16* __restrict__ h16a,          // optional f16 mirror, stride K1
                 _Float16* __restrict__ h16b,          // optional f16 mirror, stride K1
                 int K)
{
    const int ntile = blockIdx.x;   // unit group: u in [ntile*4, ntile*4+4)
    const int mtile = blockIdx.y;   // batch group: b in [mtile*16, mtile*16+16)
    const int lane  = threadIdx.x;

    v8f acc = wmma_loop(A, Bm, K, mtile, ntile, lane);

    __shared__ float tile[16][17];          // +1 pad: dodge bank conflicts
    const int l15   = lane & 15;
    const int rbase = ((lane >> 4) & 1) * 8;
#pragma unroll
    for (int r = 0; r < 8; ++r) tile[rbase + r][l15] = acc[r];
    __syncthreads();

    // 64 (b,u) pairs per tile -> 2 per lane
#pragma unroll
    for (int p = 0; p < 2; ++p) {
        int q   = lane * 2 + p;
        int row = q >> 2;                   // 0..15
        int ul  = q & 3;                    // 0..3
        int b   = mtile * 16 + row;
        int u   = ntile * 4 + ul;
        float zi = tile[row][ul * 4 + 0] + bias[u];
        float zf = tile[row][ul * 4 + 1] + bias[UNITS + u];
        float zg = tile[row][ul * 4 + 2] + bias[2 * UNITS + u];
        float zo = tile[row][ul * 4 + 3] + bias[3 * UNITS + u];
        float ig = 1.f / (1.f + expf(-zi));
        float fg = 1.f / (1.f + expf(-zf));
        float og = 1.f / (1.f + expf(-zo));
        int   s  = b * UNITS + u;
        float cn = fg * cstate[s] + ig * tanhf(zg);
        float hn = og * tanhf(cn);
        cstate[s] = cn;
        hstate[s] = hn;
        if (h16a) h16a[(size_t)b * K1 + u] = (_Float16)hn;
        if (h16b) h16b[(size_t)b * K1 + u] = (_Float16)hn;
    }
}

// ---------------------------------------------------------------------------
// Plain WMMA GEMM with f32 store (readout projection, N not interleaved)
// ---------------------------------------------------------------------------
__global__ void __launch_bounds__(32)
wmma_gemm_kernel(const _Float16* __restrict__ A,
                 const _Float16* __restrict__ Bm,
                 float* __restrict__ C,
                 int N, int K)
{
    const int ntile = blockIdx.x;
    const int mtile = blockIdx.y;
    const int lane  = threadIdx.x;
    v8f acc = wmma_loop(A, Bm, K, mtile, ntile, lane);

    const int l15  = lane & 15;
    const int hsel = (lane >> 4) & 1;
    float* crow = C + (size_t)(mtile * 16 + hsel * 8) * N + ntile * 16 + l15;
#pragma unroll
    for (int r = 0; r < 8; ++r)
        crow[(size_t)r * N] = acc[r];
}

// ---------------------------------------------------------------------------
// Weight packing: f32 -> f16, N-major / transposed, K padded, and for the
// LSTM weights the N dim is gate-interleaved:  n' = u*4 + gate.
// ---------------------------------------------------------------------------
__global__ void pack_w0_kernel(const float* __restrict__ W0,   // (IN0, 2048)
                               const float* __restrict__ U0,   // (512, 2048)
                               _Float16* __restrict__ dst)     // (2048, K0PAD)
{
    int idx = blockIdx.x * blockDim.x + threadIdx.x;
    if (idx >= N_GATE * K0PAD) return;
    int np = idx / K0PAD, k = idx % K0PAD;
    int u = np >> 2, g = np & 3;
    int n = g * UNITS + u;                  // source column (gate-blocked)
    float v = 0.f;
    if (k < IN0)              v = W0[(size_t)k * N_GATE + n];
    else if (k < IN0 + UNITS) v = U0[(size_t)(k - IN0) * N_GATE + n];
    dst[idx] = (_Float16)v;
}

__global__ void pack_w1_kernel(const float* __restrict__ W1,   // (512, 2048)
                               const float* __restrict__ U1,   // (512, 2048)
                               _Float16* __restrict__ dst)     // (2048, K1)
{
    int idx = blockIdx.x * blockDim.x + threadIdx.x;
    if (idx >= N_GATE * K1) return;
    int np = idx / K1, k = idx % K1;
    int u = np >> 2, g = np & 3;
    int n = g * UNITS + u;
    float v = (k < UNITS) ? W1[(size_t)k * N_GATE + n]
                          : U1[(size_t)(k - UNITS) * N_GATE + n];
    dst[idx] = (_Float16)v;
}

__global__ void pack_wd_kernel(const float* __restrict__ Wd,   // (1024, 46)
                               _Float16* __restrict__ dst)     // (VPAD, 1024)
{
    int idx = blockIdx.x * blockDim.x + threadIdx.x;
    if (idx >= VPAD * K1) return;
    int n = idx / K1, k = idx % K1;
    dst[idx] = (_Float16)((n < VOCAB) ? Wd[(size_t)k * VOCAB + n] : 0.f);
}

// ---------------------------------------------------------------------------
// State init: zero f32 state block and f16 activation block
// ---------------------------------------------------------------------------
__global__ void init_state_kernel(float* __restrict__ fdst, int nf,
                                  _Float16* __restrict__ hdst, int nh)
{
    int i = blockIdx.x * blockDim.x + threadIdx.x;
    if (i < nf) fdst[i] = 0.f;
    if (i < nh) hdst[i] = (_Float16)0.f;
}

// ---------------------------------------------------------------------------
// Fused: dist(t-1) = zd + bd  (-> logits row t-1)  and build LSTM0 input
// xcat(f16, 64 x K0PAD) = [h0, y_t, context, dist, h0, 0pad].
// (trailing h0 copy feeds the stacked U0 rows of the fused weight)
// ---------------------------------------------------------------------------
__global__ void build_x0_kernel(const float* __restrict__ h0,
                                const float* __restrict__ y,
                                const float* __restrict__ ctx,
                                const float* __restrict__ zd,   // (B, VPAD)
                                const float* __restrict__ bd,
                                float* __restrict__ out,
                                _Float16* __restrict__ xcat, int t)
{
    int idx = blockIdx.x * blockDim.x + threadIdx.x;
    if (idx >= B_SZ * K0PAD) return;
    int b = idx / K0PAD, k = idx % K0PAD;
    float v = 0.f;
    if (k < UNITS)              v = h0[b * UNITS + k];
    else if (k == UNITS)        v = y[b * DEC_T + t];
    else if (k < 2 * UNITS + 1) v = ctx[b * UNITS + (k - UNITS - 1)];
    else if (k < IN0) {
        int j = k - (2 * UNITS + 1);
        if (t > 0) {
            v = zd[b * VPAD + j] + bd[j];
            out[((size_t)b * DEC_T + (t - 1)) * VOCAB + j] = v;  // logits row t-1
        }
    }
    else if (k < IN0 + UNITS)   v = h0[b * UNITS + (k - IN0)];
    xcat[idx] = (_Float16)v;
}

// Final decoder-step logits (row DEC_T-1), emitted after the loop.
__global__ void last_logits_kernel(const float* __restrict__ zd,
                                   const float* __restrict__ bd,
                                   float* __restrict__ out)
{
    int idx = blockIdx.x * blockDim.x + threadIdx.x;
    if (idx >= B_SZ * VOCAB) return;
    int b = idx / VOCAB, v = idx % VOCAB;
    out[((size_t)b * DEC_T + (DEC_T - 1)) * VOCAB + v] = zd[b * VPAD + v] + bd[v];
}

// ---------------------------------------------------------------------------
// Attention (batched GEMV, keys depend on b -> VALU path, f32 softmax).
// Scores pass is wave-cooperative: each of 8 waves owns 128 encoder rows;
// the 32 lanes read consecutive float4s of one row (coalesced 512B segments),
// FMA against the register-resident query, then cross-lane shfl_xor reduce.
// Context pass: thread-per-feature column, coalesced over d.
// ---------------------------------------------------------------------------
__global__ void __launch_bounds__(256)
attention_kernel(const float* __restrict__ h,     // (B, ENC_T, UNITS)
                 const float* __restrict__ h0,    // (B, UNITS)
                 float* __restrict__ ctx,         // (B, UNITS)
                 _Float16* __restrict__ x1cat,    // (B, K1): ctx -> [0,512)
                 _Float16* __restrict__ hd)       // (B, K1): ctx -> [512,1024)
{
    const int b    = blockIdx.x;
    const int tid  = threadIdx.x;
    const int lane = tid & 31;
    const int wave = tid >> 5;               // 0..7
    __shared__ float sc[ENC_T];
    __shared__ float red[256];

    // query chunk in registers: lane covers d = j*128 + lane*4 .. +3 (float4)
    float4 qr[4];
#pragma unroll
    for (int j = 0; j < 4; ++j)
        qr[j] = *(const float4*)(h0 + b * UNITS + j * 128 + lane * 4);

    const float* __restrict__ hb = h + (size_t)b * ENC_T * UNITS;

    // ---- scores: wave w handles t in [w*128, w*128+128) ----
    for (int t = wave * 128; t < wave * 128 + 128; ++t) {
        const float* __restrict__ row = hb + (size_t)t * UNITS;
        float s = 0.f;
#pragma unroll
        for (int j = 0; j < 4; ++j) {
            float4 hv = *(const float4*)(row + j * 128 + lane * 4);
            s = fmaf(qr[j].x, hv.x, s);
            s = fmaf(qr[j].y, hv.y, s);
            s = fmaf(qr[j].z, hv.z, s);
            s = fmaf(qr[j].w, hv.w, s);
        }
#pragma unroll
        for (int off = 16; off > 0; off >>= 1)
            s += __shfl_xor(s, off, 32);
        if (lane == 0) sc[t] = s;
    }
    __syncthreads();

    // ---- softmax over sc[0..ENC_T) ----
    float m = -3.4e38f;
    for (int t = tid; t < ENC_T; t += 256) m = fmaxf(m, sc[t]);
    red[tid] = m; __syncthreads();
    for (int s = 128; s > 0; s >>= 1) {
        if (tid < s) red[tid] = fmaxf(red[tid], red[tid + s]);
        __syncthreads();
    }
    m = red[0]; __syncthreads();

    float ss = 0.f;
    for (int t = tid; t < ENC_T; t += 256) {
        float e = expf(sc[t] - m);
        sc[t] = e;
        ss += e;
    }
    red[tid] = ss; __syncthreads();
    for (int s = 128; s > 0; s >>= 1) {
        if (tid < s) red[tid] += red[tid + s];
        __syncthreads();
    }
    float inv = 1.f / red[0]; __syncthreads();
    for (int t = tid; t < ENC_T; t += 256) sc[t] *= inv;
    __syncthreads();

    // ---- context: thread owns 2 consecutive d (float2, coalesced) ----
    {
        int d = tid * 2;
        float2 acc = make_float2(0.f, 0.f);
        for (int t = 0; t < ENC_T; ++t) {
            float  w  = sc[t];
            float2 hv = *(const float2*)(hb + (size_t)t * UNITS + d);
            acc.x = fmaf(w, hv.x, acc.x);
            acc.y = fmaf(w, hv.y, acc.y);
        }
        *(float2*)(ctx + b * UNITS + d) = acc;
        _Float16 c0 = (_Float16)acc.x, c1 = (_Float16)acc.y;
        x1cat[(size_t)b * K1 + d]              = c0;  // ctx -> LSTM1 input lo
        x1cat[(size_t)b * K1 + d + 1]          = c1;
        hd[(size_t)b * K1 + UNITS + d]         = c0;  // ctx -> readout hi
        hd[(size_t)b * K1 + UNITS + d + 1]     = c1;
    }
}

// ---------------------------------------------------------------------------
// Final log_softmax over last dim (in-place on d_out), one block per row
// ---------------------------------------------------------------------------
__global__ void __launch_bounds__(64)
log_softmax_kernel(float* __restrict__ out)
{
    __shared__ float red[64];
    const int row = blockIdx.x;
    const int tid = threadIdx.x;
    float* p = out + (size_t)row * VOCAB;
    float v = (tid < VOCAB) ? p[tid] : -3.4e38f;
    red[tid] = v; __syncthreads();
    for (int s = 32; s > 0; s >>= 1) {
        if (tid < s) red[tid] = fmaxf(red[tid], red[tid + s]);
        __syncthreads();
    }
    float m = red[0]; __syncthreads();
    float e = (tid < VOCAB) ? expf(v - m) : 0.f;
    red[tid] = e; __syncthreads();
    for (int s = 32; s > 0; s >>= 1) {
        if (tid < s) red[tid] += red[tid + s];
        __syncthreads();
    }
    float lse = logf(red[0]);
    if (tid < VOCAB) p[tid] = v - m - lse;
}

// ---------------------------------------------------------------------------
// Host-side orchestration (graph-capturable: stream launches only)
// ---------------------------------------------------------------------------
extern "C" void kernel_launch(void* const* d_in, const int* in_sizes, int n_in,
                              void* d_out, int out_size, void* d_ws, size_t ws_size,
                              hipStream_t stream)
{
    (void)in_sizes; (void)n_in; (void)out_size; (void)ws_size;

    const float* h  = (const float*)d_in[0];
    const float* y  = (const float*)d_in[1];
    const float* W0 = (const float*)d_in[2];
    const float* U0 = (const float*)d_in[3];
    const float* b0 = (const float*)d_in[4];
    const float* W1 = (const float*)d_in[5];
    const float* U1 = (const float*)d_in[6];
    const float* b1 = (const float*)d_in[7];
    const float* Wd = (const float*)d_in[8];
    const float* bd = (const float*)d_in[9];
    float* out = (float*)d_out;

    char* ws = (char*)d_ws;
    // f32 state block (contiguous for one-shot zeroing)
    float*    h0    = (float*)(ws + 0);                    // 64*512
    float*    c0    = (float*)(ws + 131072);
    float*    h1    = (float*)(ws + 262144);
    float*    c1    = (float*)(ws + 393216);
    float*    ctx   = (float*)(ws + 524288);
    // f16 activation block (contiguous)
    _Float16* x1cat = (_Float16*)(ws + 655360);            // 64*1024
    _Float16* hd    = (_Float16*)(ws + 786432);            // 64*1024
    _Float16* xcat  = (_Float16*)(ws + 917504);            // 64*1600
    float*    zd    = (float*)(ws + 1122304);              // 64*48
    _Float16* w0c   = (_Float16*)(ws + 1134592);           // 2048*1600
    _Float16* w1c   = (_Float16*)(ws + 7688192);           // 2048*1024
    _Float16* wdc   = (_Float16*)(ws + 11882496);          // 48*1024

    // ---- one-time (per launch, deterministic) weight pack + state init ----
    pack_w0_kernel<<<(N_GATE * K0PAD + 255) / 256, 256, 0, stream>>>(W0, U0, w0c);
    pack_w1_kernel<<<(N_GATE * K1 + 255) / 256, 256, 0, stream>>>(W1, U1, w1c);
    pack_wd_kernel<<<(VPAD * K1 + 255) / 256, 256, 0, stream>>>(Wd, wdc);
    {
        int nf = 5 * B_SZ * UNITS;                  // h0,c0,h1,c1,ctx
        int nh = 2 * B_SZ * K1;                     // x1cat + hd
        int mx = nf > nh ? nf : nh;
        init_state_kernel<<<(mx + 255) / 256, 256, 0, stream>>>(h0, nf, x1cat, nh);
    }

    // ---- decoder recurrence: 5 kernels per step ----
    const dim3 g_gate(N_GATE / 16, B_SZ / 16);   // (128, 4)
    const dim3 g_dist(VPAD / 16, B_SZ / 16);     // (3, 4)
    for (int t = 0; t < DEC_T; ++t) {
        build_x0_kernel<<<(B_SZ * K0PAD + 255) / 256, 256, 0, stream>>>(
            h0, y, ctx, zd, bd, out, xcat, t);
        wmma_lstm_kernel<<<g_gate, 32, 0, stream>>>(
            xcat, w0c, b0, h0, c0, (_Float16*)nullptr, (_Float16*)nullptr, K0PAD);
        attention_kernel<<<B_SZ, 256, 0, stream>>>(h, h0, ctx, x1cat, hd);
        wmma_lstm_kernel<<<g_gate, 32, 0, stream>>>(
            x1cat, w1c, b1, h1, c1, x1cat + UNITS /*stride K1*/, hd /*stride K1*/, K1);
        wmma_gemm_kernel<<<g_dist, 32, 0, stream>>>(hd, wdc, zd, VPAD, K1);
    }

    // ---- tail: last logits row + log_softmax over (B*DEC_T, VOCAB) ----
    last_logits_kernel<<<(B_SZ * VOCAB + 255) / 256, 256, 0, stream>>>(zd, bd, out);
    log_softmax_kernel<<<B_SZ * DEC_T, 64, 0, stream>>>(out);
}
